// BottleneckBoxConv_46316927320139
// MI455X (gfx1250) — matrix operand order repaired and checked by reading, not compiled, not run on hardware
//
#include <hip/hip_runtime.h>
#include <hip/hip_bf16.h>

// ---------------- problem constants ----------------
#define BATCH   4
#define C_IN    128
#define NBOX    4
#define H_DIM   256
#define W_DIM   256
#define BT_DIM  32            // bottleneck channels
#define HW      (H_DIM * W_DIM)          // 65536
#define SW      272                      // padded integral-image row stride (floats)
#define SH      (H_DIM + 1)              // 257 rows
#define CHS     (SH * SW)                // floats per channel of S = 69904
#define H_ELEMS ((size_t)BATCH * BT_DIM * HW)   // 8,388,608 floats

typedef __attribute__((ext_vector_type(16))) _Float16 v16h;
typedef __attribute__((ext_vector_type(8)))  float    v8f;

// =====================================================================
// Kernel 1: 1x1 conv (x[b,c,p] . w1[k,c]) + BN1, via V_WMMA_F32_16X16X32_F16
// BN1 scale folded into f16 weights, bias added in epilogue.
// Block = 256 threads (8 waves); each wave owns a 16-pixel M-tile and
// produces both 16-channel N-tiles (BT=32) over K=128 in 4 chunks of 32.
// =====================================================================
__global__ __launch_bounds__(256) void k_conv_bn1(
    const float* __restrict__ x, const float* __restrict__ w1,
    const float* __restrict__ g1, const float* __restrict__ b1,
    const float* __restrict__ m1, const float* __restrict__ v1,
    float* __restrict__ h)
{
    __shared__ _Float16 w1s[BT_DIM][130];   // padded: bank-conflict-free column reads
    __shared__ float    hs[BT_DIM][132];    // D-matrix transpose staging
    __shared__ float    bias1s[BT_DIM];

    const int t = threadIdx.x;

    // Stage BN1-folded weights once per block (uniform across grid).
    for (int idx = t; idx < BT_DIM * C_IN; idx += 256) {
        const int n = idx >> 7, c = idx & 127;
        const float inv = g1[n] * rsqrtf(v1[n] + 1e-5f);
        w1s[n][c] = (_Float16)(w1[n * C_IN + c] * inv);
    }
    if (t < BT_DIM) {
        const float inv = g1[t] * rsqrtf(v1[t] + 1e-5f);
        bias1s[t] = b1[t] - m1[t] * inv;
    }
    __syncthreads();

    const int wave = t >> 5, lane = t & 31;
    const int b        = blockIdx.x >> 9;           // 512 blocks per batch
    const int blockPix = (blockIdx.x & 511) * 128;  // 128 pixels per block
    const int p0       = blockPix + wave * 16;      // 16-pixel M tile per wave

    const float* __restrict__ xb = x + (size_t)b * C_IN * HW + p0;

    // A-matrix f16 16x32 lane layout (ISA 7.12.2):
    //   m = lane&15 ; kbase = (lane<16)?0:8 ; half e: K = kbase + e + (e>=8 ? 8 : 0)
    const int m      = lane & 15;
    const int kbaseA = (lane & 16) ? 8 : 0;
    // B-matrix 32x16: n = lane&15 ; kbase = (lane<16)?0:8 ; half e: K = kbase + e + (e>=8 ? 8 : 0)
    const int n      = lane & 15;
    const int kbaseB = (lane & 16) ? 8 : 0;

    // Pre-gather the 8 B fragments (2 N-tiles x 4 K-chunks) — reused by every M tile.
    v16h bf[2][4];
#pragma unroll
    for (int nt = 0; nt < 2; ++nt)
#pragma unroll
        for (int kc = 0; kc < 4; ++kc) {
            v16h f;
#pragma unroll
            for (int e = 0; e < 16; ++e)
                f[e] = w1s[nt * 16 + n][kc * 32 + kbaseB + e + ((e >= 8) ? 8 : 0)];
            bf[nt][kc] = f;
        }

    v8f acc0 = {}, acc1 = {};
#pragma unroll
    for (int kc = 0; kc < 4; ++kc) {
        v16h a;
#pragma unroll
        for (int e = 0; e < 16; ++e) {
            const int ch = kc * 32 + kbaseA + e + ((e >= 8) ? 8 : 0);
            a[e] = (_Float16)xb[(size_t)ch * HW + m];
        }
        // D = A x B + C   (f32 accumulate)
        acc0 = __builtin_amdgcn_wmma_f32_16x16x32_f16(false, a, false, bf[0][kc],
                                                      (short)0, acc0, false, false);
        acc1 = __builtin_amdgcn_wmma_f32_16x16x32_f16(false, a, false, bf[1][kc],
                                                      (short)0, acc1, false, false);
    }

    // D layout: lane l, VGPR r -> M = r + 8*(l>=16), N = l&15.  Stage to LDS.
    const int   mhi = (lane >> 4) << 3;
    const float bn0 = bias1s[n], bn1v = bias1s[16 + n];
#pragma unroll
    for (int r = 0; r < 8; ++r) {
        hs[n][wave * 16 + r + mhi]      = acc0[r] + bn0;
        hs[16 + n][wave * 16 + r + mhi] = acc1[r] + bn1v;
    }
    __syncthreads();

    // Coalesced NCHW store of the 32x128 tile.
    float* __restrict__ hb = h + (size_t)b * BT_DIM * HW + blockPix;
    for (int idx = t; idx < BT_DIM * 128; idx += 256) {
        const int ch = idx >> 7, p = idx & 127;
        hb[(size_t)ch * HW + p] = hs[ch][p];
    }
}

// =====================================================================
// Kernel 2a: row prefix sums -> S[bc, i+1, 1..256]; S[bc, i+1, 0] = 0
// One wave per row (8 elems/lane serial + wave32 shfl scan).
// =====================================================================
__global__ __launch_bounds__(256) void k_rowscan(
    const float* __restrict__ h, float* __restrict__ S)
{
    const int t = threadIdx.x, wave = t >> 5, lane = t & 31;
    const int row = blockIdx.x * 8 + wave;   // 0 .. B*BT*H-1
    const int bc  = row >> 8;
    const int i   = row & 255;

    const float* __restrict__ src = h + (size_t)bc * HW + (size_t)i * W_DIM + lane * 8;
    float v[8];
#pragma unroll
    for (int q = 0; q < 8; ++q) v[q] = src[q];
#pragma unroll
    for (int q = 1; q < 8; ++q) v[q] += v[q - 1];

    float incl = v[7];
#pragma unroll
    for (int d = 1; d < 32; d <<= 1) {
        const float u = __shfl_up(incl, d, 32);
        if (lane >= d) incl += u;
    }
    const float excl = incl - v[7];

    float* __restrict__ dst = S + (size_t)bc * CHS + (size_t)(i + 1) * SW;
    if (lane == 0) dst[0] = 0.0f;
#pragma unroll
    for (int q = 0; q < 8; ++q) dst[1 + lane * 8 + q] = v[q] + excl;
}

// =====================================================================
// Kernel 2b: column prefix sums (completes the 2D integral image).
// One block per channel; thread t owns column t (t==0 also column 256).
// =====================================================================
__device__ __forceinline__ void col_scan(float* __restrict__ Sc, int c) {
    Sc[c] = 0.0f;                       // zero row 0
    float run = 0.0f;
#pragma unroll 4
    for (int i = 1; i <= H_DIM; ++i) {
        run += Sc[(size_t)i * SW + c];
        Sc[(size_t)i * SW + c] = run;
    }
}

__global__ __launch_bounds__(256) void k_colscan(float* __restrict__ S)
{
    float* __restrict__ Sc = S + (size_t)blockIdx.x * CHS;
    col_scan(Sc, threadIdx.x);
    if (threadIdx.x == 0) col_scan(Sc, W_DIM);
}

// =====================================================================
// Kernel 3: box filter (4 bilinear samples of S) + BN2 + ReLU + residual.
// blockIdx = (b, k, i); thread = j.  Row coords uniform -> SGPR; column
// coords per-lane, consecutive -> coalesced L2-resident gathers.
// =====================================================================
__global__ __launch_bounds__(256) void k_box_bn2(
    const float* __restrict__ S, const float* __restrict__ x,
    const float* __restrict__ xmn, const float* __restrict__ xmx,
    const float* __restrict__ ymn, const float* __restrict__ ymx,
    const float* __restrict__ g2, const float* __restrict__ b2,
    const float* __restrict__ m2, const float* __restrict__ v2,
    float* __restrict__ out)
{
    const int j = threadIdx.x;
    const int i = blockIdx.x & 255;
    const int k = (blockIdx.x >> 8) & 127;   // output channel = bt*NB + box = k
    const int b = blockIdx.x >> 15;
    const int bt = k >> 2;

    const size_t oidx = (((size_t)b * C_IN + k) * H_DIM + i) * W_DIM + j;
    __builtin_prefetch(x + oidx, 0, 0);      // residual read -> global_prefetch_b8

    const float a_xmn = xmn[k], a_xmx = xmx[k];
    const float a_ymn = ymn[k], a_ymx = ymx[k];
    const float inv2  = g2[k] * rsqrtf(v2[k] + 1e-3f);
    const float bias2 = b2[k] - m2[k] * inv2;

    const float* __restrict__ Sc = S + (size_t)(b * BT_DIM + bt) * CHS;

    // row coordinates (uniform across the block)
    float u_lo = fminf(fmaxf((float)i + a_xmn, 0.0f), 256.0f);
    float u_hi = fminf(fmaxf((float)i + a_xmx, 0.0f), 256.0f);
    const int i0l = min((int)u_lo, 255); const float wul = u_lo - (float)i0l;
    const int i0h = min((int)u_hi, 255); const float wuh = u_hi - (float)i0h;

    // column coordinates (per lane)
    float v_lo = fminf(fmaxf((float)j + a_ymn, 0.0f), 256.0f);
    float v_hi = fminf(fmaxf((float)j + a_ymx, 0.0f), 256.0f);
    const int j0l = min((int)v_lo, 255); const float wvl = v_lo - (float)j0l;
    const int j0h = min((int)v_hi, 255); const float wvh = v_hi - (float)j0h;

    const float* __restrict__ rA0 = Sc + (size_t)i0l * SW;
    const float* __restrict__ rA1 = rA0 + SW;
    const float* __restrict__ rB0 = Sc + (size_t)i0h * SW;
    const float* __restrict__ rB1 = rB0 + SW;

    // row-lerped samples at the 4 needed columns
    const float Al0 = rA0[j0l]     + (rA1[j0l]     - rA0[j0l])     * wul;
    const float Al1 = rA0[j0l + 1] + (rA1[j0l + 1] - rA0[j0l + 1]) * wul;
    const float Ah0 = rA0[j0h]     + (rA1[j0h]     - rA0[j0h])     * wul;
    const float Ah1 = rA0[j0h + 1] + (rA1[j0h + 1] - rA0[j0h + 1]) * wul;
    const float Bl0 = rB0[j0l]     + (rB1[j0l]     - rB0[j0l])     * wuh;
    const float Bl1 = rB0[j0l + 1] + (rB1[j0l + 1] - rB0[j0l + 1]) * wuh;
    const float Bh0 = rB0[j0h]     + (rB1[j0h]     - rB0[j0h])     * wuh;
    const float Bh1 = rB0[j0h + 1] + (rB1[j0h + 1] - rB0[j0h + 1]) * wuh;

    const float F00 = Bl0 + (Bl1 - Bl0) * wvl;   // (u_hi, v_lo) -> F10
    const float F01 = Bh0 + (Bh1 - Bh0) * wvh;   // (u_hi, v_hi) -> F11
    const float F10 = Al0 + (Al1 - Al0) * wvl;   // (u_lo, v_lo) -> F00
    const float F11 = Ah0 + (Ah1 - Ah0) * wvh;   // (u_lo, v_hi) -> F01

    const float area = (a_xmx - a_xmn) * (a_ymx - a_ymn);
    float val = (F01 - F00 - F11 + F10) / area;  // F11 - F10 - F01 + F00

    val = fmaxf(val * inv2 + bias2, 0.0f);       // BN2 + ReLU
    out[oidx] = fmaxf(x[oidx] + val, 0.0f);      // residual + ReLU
}

// =====================================================================
extern "C" void kernel_launch(void* const* d_in, const int* in_sizes, int n_in,
                              void* d_out, int out_size, void* d_ws, size_t ws_size,
                              hipStream_t stream)
{
    const float* x   = (const float*)d_in[0];
    const float* w1  = (const float*)d_in[1];
    const float* g1  = (const float*)d_in[2];
    const float* b1  = (const float*)d_in[3];
    const float* m1  = (const float*)d_in[4];
    const float* v1  = (const float*)d_in[5];
    const float* xmn = (const float*)d_in[6];
    const float* xmx = (const float*)d_in[7];
    const float* ymn = (const float*)d_in[8];
    const float* ymx = (const float*)d_in[9];
    const float* g2  = (const float*)d_in[10];
    const float* b2  = (const float*)d_in[11];
    const float* m2  = (const float*)d_in[12];
    const float* v2  = (const float*)d_in[13];
    float* out = (float*)d_out;

    float* h = (float*)d_ws;            // B*BT*H*W          = 32 MiB
    float* S = h + H_ELEMS;             // B*BT * 257*272    = 35.8 MiB

    // Stage 1: WMMA 1x1 conv + BN1
    k_conv_bn1<<<dim3(BATCH * (HW / 128)), dim3(256), 0, stream>>>(
        x, w1, g1, b1, m1, v1, h);

    // Stage 2: 2D integral image
    k_rowscan<<<dim3((BATCH * BT_DIM * H_DIM) / 8), dim3(256), 0, stream>>>(h, S);
    k_colscan<<<dim3(BATCH * BT_DIM), dim3(256), 0, stream>>>(S);

    // Stage 3: box sampling + BN2 + ReLU + residual + ReLU
    k_box_bn2<<<dim3(BATCH * C_IN * H_DIM), dim3(256), 0, stream>>>(
        S, x, xmn, xmx, ymn, ymx, g2, b2, m2, v2, out);
}
